// RNNClassifier_9363028705538
// MI455X (gfx1250) — matrix operand order repaired and checked by reading, not compiled, not run on hardware
//
#include <hip/hip_runtime.h>
#include <hip/hip_bf16.h>

// RNN classifier, fused for MI455X (gfx1250, wave32, WMMA).
//
// Plan:
//  k1: embW[V,H] = emb @ W_ih^T + b_ih + b_hh      (tiny: 57x128, in d_ws)
//  k2: per 16-row batch tile (256 WGs, 8 waves):
//        - LDS-resident: W_hh split to bf16 hi/lo, embW table, x tile, h(hi/lo)
//        - per-wave W_hh B-fragments preloaded to VGPRs (constant over t)
//        - t-loop: h' = tanh(embW[x_t] + h @ W_hh^T) via 12x v_wmma_f32_16x16x32_bf16
//          (3-term hi/lo split ~= f32 accuracy on the fast bf16 path)
//        - masked update (packed sequences); early exit at tile max length
//        - head: relu(h) @ W_out^T + b_out, log_softmax, store [B,O]

#define B_TOT 4096
#define T_LEN 512
#define VOCAB 57
#define EMB   20
#define HID   128
#define OUT_C 18
#define BM    16

typedef __attribute__((ext_vector_type(16))) __bf16       v16bf;
typedef __attribute__((ext_vector_type(8)))  float        v8f;
typedef __attribute__((ext_vector_type(8)))  unsigned int v8u;

__device__ __forceinline__ float bf2f(unsigned short h) {
  unsigned int u = ((unsigned int)h) << 16;
  return __builtin_bit_cast(float, u);
}
__device__ __forceinline__ unsigned short f2bf_rne(float f) {
  unsigned int u = __builtin_bit_cast(unsigned int, f);
  unsigned int r = u + 0x7FFFu + ((u >> 16) & 1u);
  return (unsigned short)(r >> 16);
}
__device__ __forceinline__ unsigned short f2bf_trunc(float f) {
  unsigned int u = __builtin_bit_cast(unsigned int, f);
  return (unsigned short)(u >> 16);
}
__device__ __forceinline__ v16bf make_frag(uint4 a, uint4 b) {
  v8u u;
  u[0] = a.x; u[1] = a.y; u[2] = a.z; u[3] = a.w;
  u[4] = b.x; u[5] = b.y; u[6] = b.z; u[7] = b.w;
  return __builtin_bit_cast(v16bf, u);
}

// Branch-free tanh: no EXEC churn, no IEEE-division fixup in the hot loop.
// e = exp(-2|x|) in (0,1] (never overflows), tanh(|x|) = (1-e)*rcp(1+e).
// v_exp_f32 / v_rcp_f32 are TRANS ops and co-execute with the WMMA pipe.
// |err| < ~1e-6 absolute, far below the bf16 hi/lo storage quantization of h.
__device__ __forceinline__ float fast_tanh(float x) {
  float ax = fabsf(x);
  float e  = __expf(-2.0f * ax);                       // v_exp_f32
  float r  = (1.0f - e) * __builtin_amdgcn_rcpf(1.0f + e);  // v_rcp_f32 + mul
  return copysignf(r, x);
}

// ---------------- kernel 1: projected embedding table ----------------
__global__ void embw_kernel(const float* __restrict__ emb,
                            const float* __restrict__ W_ih,
                            const float* __restrict__ b_ih,
                            const float* __restrict__ b_hh,
                            float* __restrict__ embW) {
  int i = blockIdx.x * blockDim.x + threadIdx.x;
  if (i < VOCAB * HID) {
    int v = i / HID, h = i - v * HID;
    float s = b_ih[h] + b_hh[h];
#pragma unroll
    for (int e = 0; e < EMB; ++e) s = fmaf(emb[v * EMB + e], W_ih[h * EMB + e], s);
    embW[i] = s;
  }
}

// LDS layout (dynamic, bytes):
//  w_hi   [128*128] u16 :      0 .. 32768
//  w_lo   [128*128] u16 :  32768 .. 65536
//  embW_s [57*128]  f32 :  65536 .. 94720
//  h_hi   [16*128]  u16 :  94720 .. 98816
//  h_lo   [16*128]  u16 :  98816 .. 102912
//  xbuf   [16*512]  i32 : 102912 .. 135680
//  lens   [16]      i32 : 135680 .. 135744
//  logits [16*18]   f32 : 135744 .. 136896
#define SMEM_BYTES 136896

// ---------------- kernel 2: fused recurrence + head ----------------
__global__ void __launch_bounds__(256) rnn_fused_kernel(
    const int*   __restrict__ x,
    const int*   __restrict__ x_len,
    const float* __restrict__ W_hh,
    const float* __restrict__ W_out,
    const float* __restrict__ b_out,
    const float* __restrict__ embW,
    float*       __restrict__ out) {
  extern __shared__ char smem[];
  unsigned short* w_hi   = (unsigned short*)(smem);
  unsigned short* w_lo   = (unsigned short*)(smem + 32768);
  float*          embW_s = (float*)         (smem + 65536);
  unsigned short* h_hi   = (unsigned short*)(smem + 94720);
  unsigned short* h_lo   = (unsigned short*)(smem + 98816);
  int*            xbuf   = (int*)           (smem + 102912);
  int*            lens   = (int*)           (smem + 135680);
  float*          logits = (float*)         (smem + 135744);

  const int tid = threadIdx.x;
  const int m0  = blockIdx.x * BM;

  // --- stage LDS: split W_hh into bf16 hi/lo; copy embW; zero h; stage x tile ---
  for (int i = tid; i < HID * HID; i += 256) {
    float w = W_hh[i];
    unsigned short hi = f2bf_trunc(w);
    w_hi[i] = hi;
    w_lo[i] = f2bf_rne(w - bf2f(hi));
  }
  for (int i = tid; i < VOCAB * HID; i += 256) embW_s[i] = embW[i];
  for (int i = tid; i < BM * HID; i += 256) { h_hi[i] = 0; h_lo[i] = 0; }
  for (int i = tid; i < BM * T_LEN; i += 256) {
    int m = i >> 9;             // T_LEN == 512
    int t = i & (T_LEN - 1);
    xbuf[i] = x[(m0 + m) * T_LEN + t];
  }
  if (tid < BM) lens[tid] = x_len[m0 + tid];
  __syncthreads();

  // lengths sorted descending globally -> tile max bounds the recurrence
  int tmax = 1;
#pragma unroll
  for (int m = 0; m < BM; ++m) tmax = max(tmax, lens[m]);

  const int lane = tid & 31;
  const int wave = tid >> 5;
  const int half = lane >> 4;   // which 16-lane half
  const int nl   = lane & 15;
  const int nn   = wave * 16 + nl;   // this lane's output column (N)

  // hoist per-lane loop-invariant lengths out of the recurrence (LDS -> VGPR)
  int lenr[8];
#pragma unroll
  for (int r = 0; r < 8; ++r) lenr[r] = lens[r + 8 * half];

  // --- preload this wave's W_hh B-fragments (B is 32x16 per K-chunk) ---
  // B layout (16-bit, 32x16): lanes 0-15 hold K {0-7,16-23}, lanes 16-31 K {8-15,24-31},
  // matching the A fragment K order.
  v16bf Bhi[4], Blo[4];
#pragma unroll
  for (int c = 0; c < 4; ++c) {
    int off = nn * HID + 32 * c + 8 * half;
    Bhi[c] = make_frag(*(const uint4*)(w_hi + off), *(const uint4*)(w_hi + off + 16));
    Blo[c] = make_frag(*(const uint4*)(w_lo + off), *(const uint4*)(w_lo + off + 16));
  }

  // --- recurrence ---
  for (int t = 0; t < tmax; ++t) {
    __syncthreads();                       // h writes from previous step visible
    v8f acc = {0.f, 0.f, 0.f, 0.f, 0.f, 0.f, 0.f, 0.f};
#pragma unroll
    for (int c = 0; c < 4; ++c) {
      // A layout (16-bit, 16x32): lanes 0-15 rows M=0..15 with K {0-7,16-23},
      // lanes 16-31 same rows with K {8-15,24-31}.
      int off = nl * HID + 32 * c + 8 * half;
      v16bf a_hi = make_frag(*(const uint4*)(h_hi + off), *(const uint4*)(h_hi + off + 16));
      v16bf a_lo = make_frag(*(const uint4*)(h_lo + off), *(const uint4*)(h_lo + off + 16));
      // 3-term hi/lo product: ~f32 accuracy at bf16 WMMA rate
      acc = __builtin_amdgcn_wmma_f32_16x16x32_bf16(false, a_hi, false, Bhi[c], (short)0, acc, false, false);
      acc = __builtin_amdgcn_wmma_f32_16x16x32_bf16(false, a_hi, false, Blo[c], (short)0, acc, false, false);
      acc = __builtin_amdgcn_wmma_f32_16x16x32_bf16(false, a_lo, false, Bhi[c], (short)0, acc, false, false);
    }
    __syncthreads();                       // all A reads done before h overwrite
#pragma unroll
    for (int r = 0; r < 8; ++r) {
      int m = r + 8 * half;                // D layout: (M = r + 8*half, N = lane&15)
      if (t < lenr[r]) {                   // packed-sequence mask: skip write = freeze
        int   idx = xbuf[(m << 9) + t];
        float val = fast_tanh(acc[r] + embW_s[idx * HID + nn]);
        unsigned short hv = f2bf_trunc(val);
        h_hi[m * HID + nn] = hv;
        h_lo[m * HID + nn] = f2bf_rne(val - bf2f(hv));
      }
    }
  }
  __syncthreads();

  // --- head: logits = relu(hT) @ W_out^T + b_out ---
  for (int p = tid; p < BM * OUT_C; p += 256) {
    int m = p / OUT_C, o = p - m * OUT_C;
    float s = b_out[o];
    const float* wrow = W_out + o * HID;
#pragma unroll 8
    for (int k = 0; k < HID; ++k) {
      float hv = bf2f(h_hi[m * HID + k]) + bf2f(h_lo[m * HID + k]);
      hv = fmaxf(hv, 0.0f);
      s = fmaf(hv, wrow[k], s);
    }
    logits[p] = s;
  }
  __syncthreads();

  // --- log_softmax per row ---
  if (tid < BM) {
    float mx = -3.402823466e38f;
#pragma unroll
    for (int o = 0; o < OUT_C; ++o) mx = fmaxf(mx, logits[tid * OUT_C + o]);
    float sum = 0.f;
#pragma unroll
    for (int o = 0; o < OUT_C; ++o) sum += expf(logits[tid * OUT_C + o] - mx);
    float lse = mx + logf(sum);
#pragma unroll
    for (int o = 0; o < OUT_C; ++o)
      out[(m0 + tid) * OUT_C + o] = logits[tid * OUT_C + o] - lse;
  }
}

// ---------------- launcher ----------------
extern "C" void kernel_launch(void* const* d_in, const int* in_sizes, int n_in,
                              void* d_out, int out_size, void* d_ws, size_t ws_size,
                              hipStream_t stream) {
  (void)in_sizes; (void)n_in; (void)out_size; (void)ws_size;
  const int*   x     = (const int*)  d_in[0];
  const int*   xlen  = (const int*)  d_in[1];
  const float* emb   = (const float*)d_in[2];
  const float* W_ih  = (const float*)d_in[3];
  const float* W_hh  = (const float*)d_in[4];
  const float* b_ih  = (const float*)d_in[5];
  const float* b_hh  = (const float*)d_in[6];
  const float* W_out = (const float*)d_in[7];
  const float* b_out = (const float*)d_in[8];
  float* out  = (float*)d_out;
  float* embW = (float*)d_ws;   // 57*128*4 = 29184 bytes of workspace

  embw_kernel<<<(VOCAB * HID + 255) / 256, 256, 0, stream>>>(emb, W_ih, b_ih, b_hh, embW);
  rnn_fused_kernel<<<B_TOT / BM, 256, SMEM_BYTES, stream>>>(
      x, xlen, W_hh, W_out, b_out, embW, out);
}
